// Attention_25984552141375
// MI455X (gfx1250) — matrix operand order repaired and checked by reading, not compiled, not run on hardware
//
#include <hip/hip_runtime.h>
#include <hip/hip_bf16.h>

// ---------------------------------------------------------------------------
// LeViT attention block on MI455X (gfx1250, wave32, WMMA 16x16x32 f16).
//
// B=32, N=784, DIM=512, H=8, KD=32, VD=128, QKV_OUT=1536, VAL_ATTN=1024.
// ~116 GFLOP total; attention is flash-fused so the 629MB BxHxNxN score
// tensor never touches HBM. Attention blocks are 7 waves (49 = 7x7 query
// tiles) sharing K/V chunks staged in LDS; the V chunk is staged with
// global_load_async_to_lds_b128 (ASYNCcnt path). All matmuls run through
// v_wmma_f32_16x16x32_f16 (f16 in, f32 accumulate).
// ---------------------------------------------------------------------------

typedef _Float16 v8h  __attribute__((ext_vector_type(8)));
typedef _Float16 v16h __attribute__((ext_vector_type(16)));
typedef float    v8f  __attribute__((ext_vector_type(8)));

#define CB   32
#define CN   784
#define CDIM 512
#define CH   8
#define CKD  32
#define CVD  128
#define CQKV 1536
#define CVA  1024
#define CEPS 1e-5f
#define QSCALE 0.17677669529663689f   // KD^-0.5

static __device__ __forceinline__ v16h cat8(v8h lo, v8h hi) {
  v16h r;
#pragma unroll
  for (int i = 0; i < 8; ++i) { r[i] = lo[i]; r[i + 8] = hi[i]; }
  return r;
}

static __device__ __forceinline__ v8f wmma_f16(v16h a, v16h b, v8f c) {
  // D = A(16x32 f16) x B(32x16 f16) + C(16x16 f32)
  return __builtin_amdgcn_wmma_f32_16x16x32_f16(
      /*neg_a=*/false, a, /*neg_b=*/false, b,
      /*c_mod=*/(short)0, c, /*reuse_a=*/false, /*reuse_b=*/false);
}

static __device__ __forceinline__ float rowmax16(float v) {
#pragma unroll
  for (int m = 1; m < 16; m <<= 1) v = fmaxf(v, __shfl_xor(v, m, 16));
  return v;
}
static __device__ __forceinline__ float rowsum16(float v) {
#pragma unroll
  for (int m = 1; m < 16; m <<= 1) v += __shfl_xor(v, m, 16);
  return v;
}

// ---------------------------------------------------------------------------
// Kernel 1: gather relative-position bias ab[h, idxs[n,m]] -> f16 (H,N,N)
// ---------------------------------------------------------------------------
__global__ __launch_bounds__(256) void bias_gather_kernel(
    const float* __restrict__ ab, const int* __restrict__ idxs,
    _Float16* __restrict__ bias) {
  size_t i = (size_t)blockIdx.x * 256 + threadIdx.x;
  size_t total = (size_t)CH * CN * CN;
  if (i >= total) return;
  int m = (int)(i % CN);
  size_t t = i / CN;
  int n = (int)(t % CN);
  int h = (int)(t / CN);
  bias[i] = (_Float16)ab[h * (28 * 28) + idxs[(size_t)n * CN + m]];
}

// ---------------------------------------------------------------------------
// Kernel 2: QKV GEMM (25088x512 @ 512x1536^T) + BN, scatter to q/k/vT (f16).
// Tile 64(M) x 128(N) x 32(K); 8 waves, each wave owns a 32x32 sub-tile
// (2x2 WMMA fragments). f32 operands converted to f16 on the LDS staging path.
// ---------------------------------------------------------------------------
__global__ __launch_bounds__(256) void qkv_gemm_kernel(
    const float* __restrict__ x, const float* __restrict__ w,
    const float* __restrict__ bn_g, const float* __restrict__ bn_b,
    const float* __restrict__ bn_m, const float* __restrict__ bn_v,
    _Float16* __restrict__ qh, _Float16* __restrict__ kh,
    _Float16* __restrict__ vt) {
  __shared__ __align__(16) _Float16 lds_a[64 * 40];
  __shared__ __align__(16) _Float16 lds_b[128 * 40];

  const int t    = threadIdx.x;
  const int lane = t & 31;
  const int wave = t >> 5;
  const int g    = lane >> 4;
  const int ln   = lane & 15;
  const int wm   = wave & 1;   // 2 waves along M (2*32 = 64 rows)
  const int wn   = wave >> 1;  // 4 waves along N (4*32 = 128 cols)
  const int gm0  = blockIdx.x * 64;
  const int gn0  = blockIdx.y * 128;

  v8f acc[2][2];
#pragma unroll
  for (int mi = 0; mi < 2; ++mi)
#pragma unroll
    for (int ni = 0; ni < 2; ++ni)
#pragma unroll
      for (int r = 0; r < 8; ++r) acc[mi][ni][r] = 0.0f;

  for (int k0 = 0; k0 < CDIM; k0 += 32) {
    {  // stage A: 64x32 f32 -> f16
      int row = t >> 2, cg = t & 3;
      const float* src = x + (size_t)(gm0 + row) * CDIM + k0 + cg * 8;
      if (k0 + 32 < CDIM) __builtin_prefetch(src + 32, 0, 1);
      v8h o;
#pragma unroll
      for (int i = 0; i < 8; ++i) o[i] = (_Float16)src[i];
      *(v8h*)&lds_a[row * 40 + cg * 8] = o;
    }
    {  // stage B: 128x32 f32 -> f16 (weights row-major [O,K])
      int row = t >> 1, cg = t & 1;
      const float* src = w + (size_t)(gn0 + row) * CDIM + k0 + cg * 16;
      v8h o0, o1;
#pragma unroll
      for (int i = 0; i < 8; ++i) { o0[i] = (_Float16)src[i]; o1[i] = (_Float16)src[i + 8]; }
      *(v8h*)&lds_b[row * 40 + cg * 16]     = o0;
      *(v8h*)&lds_b[row * 40 + cg * 16 + 8] = o1;
    }
    __syncthreads();

    v16h af[2], bf[2];
#pragma unroll
    for (int mi = 0; mi < 2; ++mi) {
      int m = wm * 32 + mi * 16 + ln;
      af[mi] = cat8(*(const v8h*)&lds_a[m * 40 + 8 * g],
                    *(const v8h*)&lds_a[m * 40 + 16 + 8 * g]);
    }
#pragma unroll
    for (int ni = 0; ni < 2; ++ni) {
      int n = wn * 32 + ni * 16 + ln;
      bf[ni] = cat8(*(const v8h*)&lds_b[n * 40 + 8 * g],
                    *(const v8h*)&lds_b[n * 40 + 16 + 8 * g]);
    }
#pragma unroll
    for (int mi = 0; mi < 2; ++mi)
#pragma unroll
      for (int ni = 0; ni < 2; ++ni)
        acc[mi][ni] = wmma_f16(af[mi], bf[ni], acc[mi][ni]);
    __syncthreads();
  }

  // Fused BN + scatter epilogue.  C/D layout: VGPR r -> row (r + 8*g).
#pragma unroll
  for (int ni = 0; ni < 2; ++ni) {
    int col = gn0 + wn * 32 + ni * 16 + ln;           // output channel 0..1535
    float sc = bn_g[col] * rsqrtf(bn_v[col] + CEPS);
    float mu = bn_m[col], be = bn_b[col];
    int h = col / 192, j = col % 192;
#pragma unroll
    for (int mi = 0; mi < 2; ++mi)
#pragma unroll
      for (int r = 0; r < 8; ++r) {
        int gm = gm0 + wm * 32 + mi * 16 + r + 8 * g; // flattened (b, n)
        int b  = gm / CN, n = gm % CN;
        float val = (acc[mi][ni][r] - mu) * sc + be;
        int bh = b * CH + h;
        if (j < CKD) {
          qh[((size_t)bh * CN + n) * CKD + j] = (_Float16)(val * QSCALE);
        } else if (j < 2 * CKD) {
          kh[((size_t)bh * CN + n) * CKD + (j - CKD)] = (_Float16)val;
        } else {
          // V stored transposed (B,H,VD,N) so PV B-fragments are contiguous.
          vt[((size_t)bh * CVD + (j - 2 * CKD)) * CN + n] = (_Float16)val;
        }
      }
  }
}

// ---------------------------------------------------------------------------
// Kernel 3: flash attention. 7-wave workgroups (49 = 7x7 query tiles), each
// wave owns one 16-query tile of one (b,h). Per 32-key chunk the block
// cooperatively stages K (32x32) with vector loads and V (128x32) with
// global_load_async_to_lds_b128 (ASYNCcnt), then each wave runs:
// 2x score WMMA + bias + online softmax + P relayout through its private LDS
// strip (same-wave s_wait_dscnt) + 8x PV WMMA. hardswish fused into f16 store.
// ---------------------------------------------------------------------------
__global__ __launch_bounds__(224) void attn_kernel(
    const _Float16* __restrict__ qh, const _Float16* __restrict__ kh,
    const _Float16* __restrict__ vt, const _Float16* __restrict__ biash,
    _Float16* __restrict__ oh) {
  __shared__ __align__(16) _Float16 lds_k[32 * 32];       // 2 KB: keys x kd
  __shared__ __align__(16) _Float16 lds_v[128 * 32];      // 8 KB: vd x keys
  __shared__ __align__(16) _Float16 lds_p[7 * 16 * 40];   // per-wave P strips

  const int t    = threadIdx.x;
  const int lane = t & 31;
  const int wave = t >> 5;                 // 0..6: one query tile per wave
  const int g    = lane >> 4;
  const int ln   = lane & 15;
  const int bh   = blockIdx.x / 7;
  const int tb   = blockIdx.x % 7;
  const int b    = bh >> 3;
  const int h    = bh & 7;
  const int n0   = (tb * 7 + wave) * 16;   // query tile base, always < 784

  _Float16* my_p = &lds_p[wave * (16 * 40)];

  // q A-fragment (16 queries x KD=32), held in registers for the whole pass.
  const _Float16* qbase = qh + ((size_t)bh * CN + n0 + ln) * CKD;
  const v16h qa = cat8(*(const v8h*)(qbase + 8 * g),
                       *(const v8h*)(qbase + 16 + 8 * g));

  float m_run[8], l_run[8];
  v8f oacc[8];
#pragma unroll
  for (int r = 0; r < 8; ++r) { m_run[r] = -3.0e38f; l_run[r] = 0.0f; }
#pragma unroll
  for (int j = 0; j < 8; ++j)
#pragma unroll
    for (int r = 0; r < 8; ++r) oacc[j][r] = 0.0f;

  for (int c = 0; c < 25; ++c) {           // 784 keys = 24.5 chunks of 32
    const int k0 = c * 32;

    __syncthreads();                        // prior chunk's LDS reads done
    if (t < 128) {
      // K chunk: regular 16B vector loads -> LDS [key][kd].
      {
        int key = t >> 2, cg = t & 3;
        *(v8h*)&lds_k[key * 32 + cg * 8] =
            *(const v8h*)(kh + ((size_t)bh * CN + k0 + key) * CKD + cg * 8);
      }
      // V chunk: async copy straight into LDS [vd][key] (no VGPR round-trip).
      {
        int vd = t;
#pragma unroll
        for (int cg = 0; cg < 4; ++cg) {
          uint32_t la = (uint32_t)(uintptr_t)&lds_v[vd * 32 + cg * 8];
          uint64_t ga = (uint64_t)(uintptr_t)
              (vt + ((size_t)bh * CVD + vd) * CN + k0 + cg * 8);
          asm volatile("global_load_async_to_lds_b128 %0, %1, off"
                       :: "v"(la), "v"(ga) : "memory");
        }
      }
    }
    asm volatile("s_wait_asynccnt 0x0" ::: "memory");
    __syncthreads();

    // Scores for two 16-key sub-tiles (KD = 32 = one WMMA K step).
    float s[2][8];
#pragma unroll
    for (int sub = 0; sub < 2; ++sub) {
      const int kk0 = k0 + 16 * sub;
      const _Float16* kbp = &lds_k[(16 * sub + ln) * 32 + 8 * g];
      v16h kb = cat8(*(const v8h*)kbp, *(const v8h*)(kbp + 16));
      v8f sf;
#pragma unroll
      for (int r = 0; r < 8; ++r) sf[r] = 0.0f;
      sf = wmma_f16(qa, kb, sf);
      if (kk0 < CN) {
#pragma unroll
        for (int r = 0; r < 8; ++r)
          s[sub][r] = sf[r] +
              (float)biash[(size_t)(h * CN + n0 + r + 8 * g) * CN + kk0 + ln];
      } else {
#pragma unroll
        for (int r = 0; r < 8; ++r) s[sub][r] = -3.0e38f;   // masked tail
      }
    }

    // Online softmax: each score row lives across 16 lanes of a half-wave.
#pragma unroll
    for (int r = 0; r < 8; ++r) {
      float mx = rowmax16(fmaxf(s[0][r], s[1][r]));
      float nm = fmaxf(m_run[r], mx);
      float cf = __expf(m_run[r] - nm);
      m_run[r] = nm;
      float p0 = __expf(s[0][r] - nm);
      float p1 = __expf(s[1][r] - nm);
      l_run[r] = l_run[r] * cf + rowsum16(p0 + p1);
#pragma unroll
      for (int j = 0; j < 8; ++j) oacc[j][r] *= cf;
      my_p[(r + 8 * g) * 40 + ln]      = (_Float16)p0;
      my_p[(r + 8 * g) * 40 + 16 + ln] = (_Float16)p1;
    }

    // Same-wave DS ordering for the cross-lane P relayout (D -> A layout).
    asm volatile("s_wait_dscnt 0x0" ::: "memory");

    const _Float16* pbase = &my_p[ln * 40];
    const v16h pa = cat8(*(const v8h*)(pbase + 8 * g),
                         *(const v8h*)(pbase + 16 + 8 * g));
#pragma unroll
    for (int j = 0; j < 8; ++j) {
      const _Float16* vbp = &lds_v[(16 * j + ln) * 32 + 8 * g];
      v16h vf = cat8(*(const v8h*)vbp, *(const v8h*)(vbp + 16));
      oacc[j] = wmma_f16(pa, vf, oacc[j]);
    }
  }

  // Normalize + hardswish + f16 store (layout (B*N, VAL_ATTN) for proj GEMM).
#pragma unroll
  for (int r = 0; r < 8; ++r) {
    float inv = 1.0f / l_run[r];
    int n = n0 + r + 8 * g;
    size_t orow = ((size_t)b * CN + n) * CVA + h * CVD;
#pragma unroll
    for (int j = 0; j < 8; ++j) {
      float xv = oacc[j][r] * inv;
      float hs = xv * fminf(fmaxf(xv + 3.0f, 0.0f), 6.0f) * (1.0f / 6.0f);
      oh[orow + 16 * j + ln] = (_Float16)hs;
    }
  }
}

// ---------------------------------------------------------------------------
// Kernel 4: proj GEMM (25088x1024 f16 @ 1024x512^T) + BN -> f32 out.
// ---------------------------------------------------------------------------
__global__ __launch_bounds__(256) void proj_gemm_kernel(
    const _Float16* __restrict__ a, const float* __restrict__ w,
    const float* __restrict__ bn_g, const float* __restrict__ bn_b,
    const float* __restrict__ bn_m, const float* __restrict__ bn_v,
    float* __restrict__ out) {
  __shared__ __align__(16) _Float16 lds_a[64 * 40];
  __shared__ __align__(16) _Float16 lds_b[128 * 40];

  const int t    = threadIdx.x;
  const int lane = t & 31;
  const int wave = t >> 5;
  const int g    = lane >> 4;
  const int ln   = lane & 15;
  const int wm   = wave & 1;
  const int wn   = wave >> 1;
  const int gm0  = blockIdx.x * 64;
  const int gn0  = blockIdx.y * 128;

  v8f acc[2][2];
#pragma unroll
  for (int mi = 0; mi < 2; ++mi)
#pragma unroll
    for (int ni = 0; ni < 2; ++ni)
#pragma unroll
      for (int r = 0; r < 8; ++r) acc[mi][ni][r] = 0.0f;

  for (int k0 = 0; k0 < CVA; k0 += 32) {
    {  // stage A: already f16, straight 16B copies
      int row = t >> 2, cg = t & 3;
      const _Float16* src = a + (size_t)(gm0 + row) * CVA + k0 + cg * 8;
      if (k0 + 32 < CVA) __builtin_prefetch(src + 32, 0, 1);
      *(v8h*)&lds_a[row * 40 + cg * 8] = *(const v8h*)src;
    }
    {  // stage B: 128x32 f32 -> f16
      int row = t >> 1, cg = t & 1;
      const float* src = w + (size_t)(gn0 + row) * CVA + k0 + cg * 16;
      v8h o0, o1;
#pragma unroll
      for (int i = 0; i < 8; ++i) { o0[i] = (_Float16)src[i]; o1[i] = (_Float16)src[i + 8]; }
      *(v8h*)&lds_b[row * 40 + cg * 16]     = o0;
      *(v8h*)&lds_b[row * 40 + cg * 16 + 8] = o1;
    }
    __syncthreads();

    v16h af[2], bf[2];
#pragma unroll
    for (int mi = 0; mi < 2; ++mi) {
      int m = wm * 32 + mi * 16 + ln;
      af[mi] = cat8(*(const v8h*)&lds_a[m * 40 + 8 * g],
                    *(const v8h*)&lds_a[m * 40 + 16 + 8 * g]);
    }
#pragma unroll
    for (int ni = 0; ni < 2; ++ni) {
      int n = wn * 32 + ni * 16 + ln;
      bf[ni] = cat8(*(const v8h*)&lds_b[n * 40 + 8 * g],
                    *(const v8h*)&lds_b[n * 40 + 16 + 8 * g]);
    }
#pragma unroll
    for (int mi = 0; mi < 2; ++mi)
#pragma unroll
      for (int ni = 0; ni < 2; ++ni)
        acc[mi][ni] = wmma_f16(af[mi], bf[ni], acc[mi][ni]);
    __syncthreads();
  }

#pragma unroll
  for (int ni = 0; ni < 2; ++ni) {
    int col = gn0 + wn * 32 + ni * 16 + ln;           // 0..511
    float sc = bn_g[col] * rsqrtf(bn_v[col] + CEPS);
    float mu = bn_m[col], be = bn_b[col];
#pragma unroll
    for (int mi = 0; mi < 2; ++mi)
#pragma unroll
      for (int r = 0; r < 8; ++r) {
        int gm = gm0 + wm * 32 + mi * 16 + r + 8 * g;
        out[(size_t)gm * CDIM + col] = (acc[mi][ni][r] - mu) * sc + be;
      }
  }
}

// ---------------------------------------------------------------------------
// Host-side launcher.
// ---------------------------------------------------------------------------
extern "C" void kernel_launch(void* const* d_in, const int* in_sizes, int n_in,
                              void* d_out, int out_size, void* d_ws, size_t ws_size,
                              hipStream_t stream) {
  const float* x      = (const float*)d_in[0];
  const float* qkv_w  = (const float*)d_in[1];
  const float* qkv_g  = (const float*)d_in[2];
  const float* qkv_b  = (const float*)d_in[3];
  const float* qkv_m  = (const float*)d_in[4];
  const float* qkv_v  = (const float*)d_in[5];
  const float* ab     = (const float*)d_in[6];
  const float* proj_w = (const float*)d_in[7];
  const float* proj_g = (const float*)d_in[8];
  const float* proj_b = (const float*)d_in[9];
  const float* proj_m = (const float*)d_in[10];
  const float* proj_v = (const float*)d_in[11];
  const int*   idxs   = (const int*)d_in[12];
  float* out = (float*)d_out;

  // Workspace carve-up (f16 scratch), ~138 MB total.
  char* wsp = (char*)d_ws;
  _Float16* q_h    = (_Float16*)wsp; wsp += (size_t)CB * CH * CN * CKD * 2;          // 12.8 MB
  _Float16* k_h    = (_Float16*)wsp; wsp += (size_t)CB * CH * CN * CKD * 2 + 2048;   // 12.8 MB (+pad: tail chunk over-read)
  _Float16* v_t    = (_Float16*)wsp; wsp += (size_t)CB * CH * CVD * CN * 2 + 256;    // 51.4 MB (+pad: tail chunk over-read)
  _Float16* o_h    = (_Float16*)wsp; wsp += (size_t)CB * CN * CVA * 2;               // 51.4 MB
  _Float16* bias_h = (_Float16*)wsp; wsp += (size_t)CH * CN * CN * 2;                // 9.8 MB

  // 1) bias gather
  {
    size_t total = (size_t)CH * CN * CN;
    int blocks = (int)((total + 255) / 256);
    bias_gather_kernel<<<blocks, 256, 0, stream>>>(ab, idxs, bias_h);
  }
  // 2) QKV GEMM + BN + scatter   (M=25088 -> 392 tiles, N=1536 -> 12 tiles)
  qkv_gemm_kernel<<<dim3(392, 12), 256, 0, stream>>>(
      x, qkv_w, qkv_g, qkv_b, qkv_m, qkv_v, q_h, k_h, v_t);
  // 3) flash attention           (256 (b,h) x 7 blocks, 7 waves each)
  attn_kernel<<<CB * CH * 7, 224, 0, stream>>>(q_h, k_h, v_t, bias_h, o_h);
  // 4) proj GEMM + BN -> out     (M=25088 -> 392 tiles, N=512 -> 4 tiles)
  proj_gemm_kernel<<<dim3(392, 4), 256, 0, stream>>>(
      o_h, proj_w, proj_g, proj_b, proj_m, proj_v, out);
}
